// DecoderBlock_45028437131954
// MI455X (gfx1250) — matrix operand order repaired and checked by reading, not compiled, not run on hardware
//
#include <hip/hip_runtime.h>
#include <hip/hip_bf16.h>
#include <math.h>

#define S_LEN 2048
#define BATCH 2
#define DM    512
#define NH    8
#define DK    64
#define DFF   2048
#define TOKENS (S_LEN*BATCH)

typedef __attribute__((ext_vector_type(16))) __bf16       bf16x16;
typedef __attribute__((ext_vector_type(8)))  float        floatx8;
typedef __attribute__((ext_vector_type(4)))  unsigned int uintx4;

// types for the async global->LDS builtin (per hipcc diagnostic: v4i pointers)
typedef int v4i_vs __attribute__((vector_size(16)));
typedef __attribute__((address_space(1))) v4i_vs* gas_v4i;
typedef __attribute__((address_space(3))) v4i_vs* lds_v4i;

union BFrag { bf16x16 f; uintx4 u[2]; };

__device__ __forceinline__ unsigned short f2bf(float x) {
  union { float f; unsigned int u; } v; v.f = x;
  unsigned int r = v.u + 0x7FFFu + ((v.u >> 16) & 1u);  // round-to-nearest-even
  return (unsigned short)(r >> 16);
}

__device__ __forceinline__ floatx8 wmma_bf16(const BFrag& a, const BFrag& b, floatx8 c) {
  return __builtin_amdgcn_wmma_f32_16x16x32_bf16(false, a.f, false, b.f,
                                                 (short)0, c, false, false);
}

// ---- async global->LDS 16B copy (CDNA5 ASYNCcnt path) with sync fallback ----
__device__ __forceinline__ void async_copy_b128(const void* g, void* l) {
#if __has_builtin(__builtin_amdgcn_global_load_async_to_lds_b128)
  v4i_vs* gg = (v4i_vs*)const_cast<void*>(g);
  v4i_vs* ll = (v4i_vs*)l;
  __builtin_amdgcn_global_load_async_to_lds_b128((gas_v4i)gg, (lds_v4i)ll, 0, 0);
#else
  *(uintx4*)l = *(const uintx4*)g;
#endif
}
__device__ __forceinline__ void wait_async_le1() {
#if __has_builtin(__builtin_amdgcn_s_wait_asynccnt)
  __builtin_amdgcn_s_wait_asynccnt(1);
#else
  asm volatile("s_wait_asynccnt 0x1" ::: "memory");
#endif
}
__device__ __forceinline__ void wait_async_le0() {
#if __has_builtin(__builtin_amdgcn_s_wait_asynccnt)
  __builtin_amdgcn_s_wait_asynccnt(0);
#else
  asm volatile("s_wait_asynccnt 0x0" ::: "memory");
#endif
}

// ---------------- elementwise f32 -> bf16 ----------------
__global__ void k_f32_to_bf16(const float* __restrict__ in,
                              unsigned short* __restrict__ out, int n) {
  int i = blockIdx.x * blockDim.x + threadIdx.x;
  if (i < n) out[i] = f2bf(in[i]);
}

// ---------------- LayerNorm: one wave per token row (D=512) ----------------
__global__ __launch_bounds__(256) void k_layernorm_bf16(
    const float* __restrict__ x, const float* __restrict__ g,
    const float* __restrict__ beta, unsigned short* __restrict__ out) {
  const int lane = threadIdx.x & 31;
  const int wave = threadIdx.x >> 5;
  const int row  = blockIdx.x * 8 + wave;
  const float* xr = x + (size_t)row * DM;
  float v[16], s = 0.f, s2 = 0.f;
#pragma unroll
  for (int i = 0; i < 16; ++i) {
    float t = xr[lane + 32 * i];
    v[i] = t; s += t; s2 += t * t;
  }
#pragma unroll
  for (int off = 16; off >= 1; off >>= 1) {
    s  += __shfl_xor(s,  off, 32);
    s2 += __shfl_xor(s2, off, 32);
  }
  const float mean = s * (1.f / DM);
  const float var  = s2 * (1.f / DM) - mean * mean;
  const float rstd = rsqrtf(var + 1e-5f);
  unsigned short* orow = out + (size_t)row * DM;
#pragma unroll
  for (int i = 0; i < 16; ++i) {
    int c = lane + 32 * i;
    orow[c] = f2bf((v[i] - mean) * rstd * g[c] + beta[c]);
  }
}

// --- pipelined GEMM: C[M,N] = A[M,K] * W[N,K]^T (+bias,relu,resid) ---
// Block = 8 waves = 128(M) x 64(N) tile. B tile (64x32) staged in LDS via
// async copies, ping/pong buffers with compile-time indices; A fragments in
// named registers (a0/a1) so nothing spills to LDS/scratch.
__global__ __launch_bounds__(256) void k_gemm_bf16(
    const unsigned short* __restrict__ A,   // [M,K] bf16 row-major
    const unsigned short* __restrict__ W,   // [N,K] bf16 row-major (torch)
    const float* __restrict__ bias,         // [N] or null
    const float* __restrict__ resid,        // [M,N] or null
    float* __restrict__ outF,               // [M,N] or null
    unsigned short* __restrict__ outB,      // [M,N] or null
    int M, int N, int K, int relu) {
  __shared__ __align__(16) unsigned short ldsB[2][64][32];
  const int tid  = threadIdx.x;
  const int lane = tid & 31;
  const int wave = tid >> 5;
  const int m0   = blockIdx.x * 128 + wave * 16;
  const int n0   = blockIdx.y * 64;
  const int half = lane >> 4;
  const int l15  = lane & 15;
  const int nk   = K >> 5;                 // 16 or 64: always even

  // this thread's 16B slice of the 64x32 B tile
  const int cn = tid >> 2;                 // 0..63 (local col)
  const int ck = (tid & 3) * 8;            // 0,8,16,24 (K chunk)
  const unsigned short* wsrc = W + (size_t)(n0 + cn) * K + ck;

  // stage tile 0
  async_copy_b128(wsrc, &ldsB[0][cn][ck]);

  const unsigned short* arow = A + (size_t)(m0 + l15) * K;
  BFrag a0, a1;
  a0.u[0] = *(const uintx4*)(arow + half * 8);
  a0.u[1] = *(const uintx4*)(arow + 16 + half * 8);

  floatx8 acc[4] = {};
  for (int kt = 0; kt < nk; kt += 2) {
    // ---- phase 0: compute tile kt from ldsB[0]; prefetch kt+1 -> ldsB[1]
    {
      async_copy_b128(wsrc + (size_t)(kt + 1) * 32, &ldsB[1][cn][ck]);
      const unsigned short* ap = arow + (size_t)(kt + 1) * 32;
      a1.u[0] = *(const uintx4*)(ap + half * 8);
      a1.u[1] = *(const uintx4*)(ap + 16 + half * 8);
      wait_async_le1();                    // tile kt resident, kt+1 in flight
    }
    __syncthreads();
    {
      BFrag bb[4];
#pragma unroll
      for (int nt = 0; nt < 4; ++nt) {
        bb[nt].u[0] = *(const uintx4*)&ldsB[0][nt * 16 + l15][half * 8];
        bb[nt].u[1] = *(const uintx4*)&ldsB[0][nt * 16 + l15][16 + half * 8];
      }
#pragma unroll
      for (int nt = 0; nt < 4; ++nt)
        acc[nt] = wmma_bf16(a0, bb[nt], acc[nt]);
    }
    __syncthreads();
    // ---- phase 1: compute tile kt+1 from ldsB[1]; prefetch kt+2 -> ldsB[0]
    if (kt + 2 < nk) {                     // uniform branch: EXEC stays full
      async_copy_b128(wsrc + (size_t)(kt + 2) * 32, &ldsB[0][cn][ck]);
      const unsigned short* ap = arow + (size_t)(kt + 2) * 32;
      a0.u[0] = *(const uintx4*)(ap + half * 8);
      a0.u[1] = *(const uintx4*)(ap + 16 + half * 8);
      wait_async_le1();
    } else {
      wait_async_le0();
    }
    __syncthreads();
    {
      BFrag bb[4];
#pragma unroll
      for (int nt = 0; nt < 4; ++nt) {
        bb[nt].u[0] = *(const uintx4*)&ldsB[1][nt * 16 + l15][half * 8];
        bb[nt].u[1] = *(const uintx4*)&ldsB[1][nt * 16 + l15][16 + half * 8];
      }
#pragma unroll
      for (int nt = 0; nt < 4; ++nt)
        acc[nt] = wmma_bf16(a1, bb[nt], acc[nt]);
    }
    __syncthreads();
  }
#pragma unroll
  for (int nt = 0; nt < 4; ++nt) {
    const int col = n0 + nt * 16 + l15;
    const float bv = bias ? bias[col] : 0.f;
#pragma unroll
    for (int r = 0; r < 8; ++r) {
      const int m = m0 + r + half * 8;
      float val = acc[nt][r] + bv;
      if (relu)  val = fmaxf(val, 0.f);
      if (resid) val += resid[(size_t)m * N + col];
      if (outF)  outF[(size_t)m * N + col] = val;
      if (outB)  outB[(size_t)m * N + col] = f2bf(val);
    }
  }
}

// ---------------- V transpose: [token, DM] -> [B,H,DK,S] ----------------
__global__ void k_transpose_v(const unsigned short* __restrict__ V,
                              unsigned short* __restrict__ Vt) {
  int i = blockIdx.x * blockDim.x + threadIdx.x;
  if (i >= BATCH * NH * DK * S_LEN) return;
  int s = i % S_LEN;
  int d = (i / S_LEN) % DK;
  int h = (i / (S_LEN * DK)) % NH;
  int b = i / (S_LEN * DK * NH);
  Vt[i] = V[(size_t)(s * BATCH + b) * DM + h * DK + d];
}

// ---------------- flash attention: one wave = 16 query rows, dk=64 ----------------
__global__ __launch_bounds__(256) void k_attention(
    const unsigned short* __restrict__ Q,    // [T, DM] token-major
    const unsigned short* __restrict__ Km,   // [T, DM]
    const unsigned short* __restrict__ Vt,   // [B,H,DK,S]
    const unsigned char* __restrict__ kpad,  // [B,S] or null (1 = masked)
    unsigned short* __restrict__ O,          // [T, DM]
    int causal) {
  __shared__ __align__(16) unsigned short pbuf[8][16 * 32];
  const int lane = threadIdx.x & 31;
  const int wave = threadIdx.x >> 5;
  const int wid  = blockIdx.x * 8 + wave;
  const int nqt  = S_LEN / 16;
  const int qt   = wid % nqt;
  const int h    = (wid / nqt) % NH;
  const int b    = wid / (nqt * NH);
  const int q0   = qt * 16;
  const int half = lane >> 4;
  const int l15  = lane & 15;
  const float NEGINF = -__builtin_inff();

  // preload Q fragments (16 rows x dk=64 -> two 16x32 A-frags)
  const unsigned short* qrow =
      Q + (size_t)((q0 + l15) * BATCH + b) * DM + h * DK;
  BFrag qa[2];
#pragma unroll
  for (int kk = 0; kk < 2; ++kk) {
    qa[kk].u[0] = *(const uintx4*)(qrow + kk * 32 + half * 8);
    qa[kk].u[1] = *(const uintx4*)(qrow + kk * 32 + 16 + half * 8);
  }

  floatx8 acc[4] = {};
  float mst[8], lst[8];
#pragma unroll
  for (int r = 0; r < 8; ++r) { mst[r] = NEGINF; lst[r] = 0.f; }

  const int kend = causal ? (q0 + 16) : S_LEN;
  for (int kb = 0; kb < kend; kb += 32) {
    floatx8 s01[2] = {};
#pragma unroll
    for (int nt = 0; nt < 2; ++nt) {           // S = Q * K^T
      const unsigned short* krow =
          Km + (size_t)((kb + nt * 16 + l15) * BATCH + b) * DM + h * DK;
#pragma unroll
      for (int kk = 0; kk < 2; ++kk) {
        BFrag bf;
        bf.u[0] = *(const uintx4*)(krow + kk * 32 + half * 8);
        bf.u[1] = *(const uintx4*)(krow + kk * 32 + 16 + half * 8);
        s01[nt] = wmma_bf16(qa[kk], bf, s01[nt]);
      }
    }
    // hoisted V-fragment loads: independent of softmax -> overlap exp/shuffles
    BFrag vf[4];
#pragma unroll
    for (int nt = 0; nt < 4; ++nt) {
      const unsigned short* vrow =
          Vt + (size_t)((b * NH + h) * DK + nt * 16 + l15) * S_LEN + kb + half * 8;
      vf[nt].u[0] = *(const uintx4*)(vrow);
      vf[nt].u[1] = *(const uintx4*)(vrow + 16);
    }
    // scale 1/sqrt(64) + masking
#pragma unroll
    for (int nt = 0; nt < 2; ++nt) {
      const int key = kb + nt * 16 + l15;
      const bool pad = kpad && kpad[b * S_LEN + key];
#pragma unroll
      for (int r = 0; r < 8; ++r) {
        const int qi = q0 + r + half * 8;
        float sv = s01[nt][r] * 0.125f;
        if ((causal && key > qi) || pad) sv = NEGINF;
        s01[nt][r] = sv;
      }
    }
    // online softmax (row lives in 16 lanes of one half-wave)
#pragma unroll
    for (int r = 0; r < 8; ++r) {
      float mloc = fmaxf(s01[0][r], s01[1][r]);
#pragma unroll
      for (int off = 8; off >= 1; off >>= 1)
        mloc = fmaxf(mloc, __shfl_xor(mloc, off, 32));
      const float mnew  = fmaxf(mst[r], mloc);
      const float alpha = (mnew == NEGINF) ? 1.f : __expf(mst[r] - mnew);
      const float p0 = (s01[0][r] == NEGINF) ? 0.f : __expf(s01[0][r] - mnew);
      const float p1 = (s01[1][r] == NEGINF) ? 0.f : __expf(s01[1][r] - mnew);
      float rs = p0 + p1;
#pragma unroll
      for (int off = 8; off >= 1; off >>= 1)
        rs += __shfl_xor(rs, off, 32);
      lst[r] = lst[r] * alpha + rs;
      mst[r] = mnew;
      s01[0][r] = p0; s01[1][r] = p1;
#pragma unroll
      for (int nt = 0; nt < 4; ++nt) acc[nt][r] *= alpha;
    }
    // P: C-layout -> A-layout via per-wave LDS bounce (no barrier needed)
#pragma unroll
    for (int nt = 0; nt < 2; ++nt)
#pragma unroll
      for (int r = 0; r < 8; ++r)
        pbuf[wave][(r + half * 8) * 32 + nt * 16 + l15] = f2bf(s01[nt][r]);
    BFrag pf;
    pf.u[0] = *(const uintx4*)&pbuf[wave][l15 * 32 + half * 8];
    pf.u[1] = *(const uintx4*)&pbuf[wave][l15 * 32 + 16 + half * 8];
    // O += P * V (V fragments already resident)
#pragma unroll
    for (int nt = 0; nt < 4; ++nt)
      acc[nt] = wmma_bf16(pf, vf[nt], acc[nt]);
  }
  // normalize + store token-major bf16 (feeds output projection GEMM)
#pragma unroll
  for (int r = 0; r < 8; ++r) {
    const float linv = (lst[r] > 0.f) ? (1.f / lst[r]) : 0.f;
    const size_t trow = (size_t)((q0 + r + half * 8) * BATCH + b) * DM + h * DK;
#pragma unroll
    for (int nt = 0; nt < 4; ++nt)
      O[trow + nt * 16 + l15] = f2bf(acc[nt][r] * linv);
  }
}

extern "C" void kernel_launch(void* const* d_in, const int* in_sizes, int n_in,
                              void* d_out, int out_size, void* d_ws, size_t ws_size,
                              hipStream_t stream) {
  const float* x      = (const float*)d_in[0];
  const float* memory = (const float*)d_in[1];
  // d_in[2] tgt_mask: causal, computed analytically in-kernel
  const unsigned char* memory_mask = (const unsigned char*)d_in[3];
  const float* sa_W = (const float*)d_in[4];
  const float* sa_b = (const float*)d_in[5];
  const float* ca_W = (const float*)d_in[6];
  const float* ca_b = (const float*)d_in[7];
  const float* w1   = (const float*)d_in[8];
  const float* b1   = (const float*)d_in[9];
  const float* w2   = (const float*)d_in[10];
  const float* b2   = (const float*)d_in[11];
  const float* ln_g = (const float*)d_in[12];
  const float* ln_b = (const float*)d_in[13];
  float* out = (float*)d_out;

  char* ws = (char*)d_ws;
  size_t off = 0;
  auto alloc = [&](size_t bytes) {
    char* p = ws + off;
    off = (off + bytes + 255) & ~(size_t)255;
    return p;
  };
  unsigned short* saWb = (unsigned short*)alloc((size_t)4 * DM * DM * 2);
  unsigned short* caWb = (unsigned short*)alloc((size_t)4 * DM * DM * 2);
  unsigned short* w1b  = (unsigned short*)alloc((size_t)DFF * DM * 2);
  unsigned short* w2b  = (unsigned short*)alloc((size_t)DM * DFF * 2);
  unsigned short* memb = (unsigned short*)alloc((size_t)TOKENS * DM * 2);
  unsigned short* nx   = (unsigned short*)alloc((size_t)TOKENS * DM * 2);
  unsigned short* Qb   = (unsigned short*)alloc((size_t)TOKENS * DM * 2);
  unsigned short* Kb   = (unsigned short*)alloc((size_t)TOKENS * DM * 2);
  unsigned short* Vb   = (unsigned short*)alloc((size_t)TOKENS * DM * 2);
  unsigned short* Vt   = (unsigned short*)alloc((size_t)TOKENS * DM * 2);
  unsigned short* AO   = (unsigned short*)alloc((size_t)TOKENS * DM * 2);
  float* x1 = (float*)alloc((size_t)TOKENS * DM * 4);
  float* x2 = (float*)alloc((size_t)TOKENS * DM * 4);
  unsigned short* hb = (unsigned short*)alloc((size_t)TOKENS * DFF * 2);

  auto cvt = [&](const float* src, unsigned short* dst, int n) {
    k_f32_to_bf16<<<(n + 255) / 256, 256, 0, stream>>>(src, dst, n);
  };
  auto gemm = [&](const unsigned short* A, const unsigned short* W,
                  const float* bias, const float* resid, float* oF,
                  unsigned short* oB, int M, int N, int K, int relu) {
    dim3 grid(M / 128, N / 64);
    k_gemm_bf16<<<grid, 256, 0, stream>>>(A, W, bias, resid, oF, oB, M, N, K, relu);
  };

  // one-time bf16 conversions (weights + memory)
  cvt(sa_W, saWb, 4 * DM * DM);
  cvt(ca_W, caWb, 4 * DM * DM);
  cvt(w1, w1b, DFF * DM);
  cvt(w2, w2b, DM * DFF);
  cvt(memory, memb, TOKENS * DM);

  const int vt_n = BATCH * NH * DK * S_LEN;

  // ---- self-attention block ----
  k_layernorm_bf16<<<TOKENS / 8, 256, 0, stream>>>(x, ln_g, ln_b, nx);
  gemm(nx, saWb + 0 * DM * DM, sa_b + 0 * DM, nullptr, nullptr, Qb, TOKENS, DM, DM, 0);
  gemm(nx, saWb + 1 * DM * DM, sa_b + 1 * DM, nullptr, nullptr, Kb, TOKENS, DM, DM, 0);
  gemm(nx, saWb + 2 * DM * DM, sa_b + 2 * DM, nullptr, nullptr, Vb, TOKENS, DM, DM, 0);
  k_transpose_v<<<(vt_n + 255) / 256, 256, 0, stream>>>(Vb, Vt);
  k_attention<<<256, 256, 0, stream>>>(Qb, Kb, Vt, nullptr, AO, 1);
  gemm(AO, saWb + 3 * DM * DM, sa_b + 3 * DM, x, x1, nullptr, TOKENS, DM, DM, 0);

  // ---- cross-attention block ----
  k_layernorm_bf16<<<TOKENS / 8, 256, 0, stream>>>(x1, ln_g + DM, ln_b + DM, nx);
  gemm(nx,   caWb + 0 * DM * DM, ca_b + 0 * DM, nullptr, nullptr, Qb, TOKENS, DM, DM, 0);
  gemm(memb, caWb + 1 * DM * DM, ca_b + 1 * DM, nullptr, nullptr, Kb, TOKENS, DM, DM, 0);
  gemm(memb, caWb + 2 * DM * DM, ca_b + 2 * DM, nullptr, nullptr, Vb, TOKENS, DM, DM, 0);
  k_transpose_v<<<(vt_n + 255) / 256, 256, 0, stream>>>(Vb, Vt);
  k_attention<<<256, 256, 0, stream>>>(Qb, Kb, Vt, memory_mask, AO, 0);
  gemm(AO, caWb + 3 * DM * DM, ca_b + 3 * DM, x1, x2, nullptr, TOKENS, DM, DM, 0);

  // ---- FFN block ----
  k_layernorm_bf16<<<TOKENS / 8, 256, 0, stream>>>(x2, ln_g + 2 * DM, ln_b + 2 * DM, nx);
  gemm(nx, w1b, b1, nullptr, nullptr, hb, TOKENS, DFF, DM, 1);
  gemm(hb, w2b, b2, x2, out, nullptr, TOKENS, DM, DFF, 0);
}
